// MMINet_66623532695763
// MI455X (gfx1250) — matrix operand order repaired and checked by reading, not compiled, not run on hardware
//
#include <hip/hip_runtime.h>
#include <hip/hip_bf16.h>

typedef __attribute__((ext_vector_type(16))) _Float16 v16h;
typedef __attribute__((ext_vector_type(8)))  _Float16 v8h;
typedef __attribute__((ext_vector_type(8)))  float    v8f;

#define BSZ   4096
#define SEQ   1024
#define HID   30
#define OPN   30
#define NVAR  4
#define NINIT 16
#define SOUT  (SEQ - NINIT)

#define N00 0.0f
#define N01 1.0f
#define N40 0.0f
#define N41 1.0f

__device__ __forceinline__ float sigf(float z) {
  return __builtin_amdgcn_rcpf(1.0f + __expf(-z));
}

// One wave32 per block; each wave owns 16 batch rows.
// Lane l: r = l&15 (hysteresis row / A-row / B-columns j0=r, j1=16+r),
//         hi = l>>4 (K-halves / row-halves per WMMA register layout).
__global__ __launch_bounds__(32)
void mmi_recurrent(const float* __restrict__ x,    const float* __restrict__ var,
                   const float* __restrict__ s0,   const float* __restrict__ Wd1,
                   const float* __restrict__ bd1,  const float* __restrict__ Wx2h,
                   const float* __restrict__ Wh2h, const float* __restrict__ Wd2,
                   const float* __restrict__ bd2,  float* __restrict__ Hout)
{
  const int lane = threadIdx.x;
  const int r    = lane & 15;
  const int hi   = lane >> 4;
  const int base = blockIdx.x * 16;
  const int b_r  = base + r;
  const int j0   = r;
  const int j1   = 16 + r;
  const bool ok1 = (j1 < HID);          // lanes r=14,15 have no valid tile1 column

  // Row 16 of Abuf is a write-dump for invalid tile1 columns (branch-free store-back).
  __shared__ __align__(32) _Float16 Abuf[17][32];
  __shared__ float hh[16], xs0s[16], xs2s[16], xs3s[16];
  __shared__ float varS[16][4];

  _Float16* const abase = &Abuf[0][0];
  const int t1off = ok1 ? j1 : (512 + lane);   // dump slots 512+lane are unique per lane
  const int t1str = ok1 ? 32 : 0;

  const float bd1_0 = bd1[0];
  const float bd2_0 = bd2[0];

  // ---- per-row var + fixed part of hyst dot ----
  const float va0 = var[b_r*NVAR+0], va1 = var[b_r*NVAR+1];
  const float va2 = var[b_r*NVAR+2], va3 = var[b_r*NVAR+3];
  const float vdot1 = va0*Wd1[OPN+0] + va1*Wd1[OPN+1] + va2*Wd1[OPN+2] + va3*Wd1[OPN+3];

  // ---- hysteresis: lane pair splits 30 elements (15 each) ----
  float w1l[15], rkl[15];
  #pragma unroll
  for (int kk = 0; kk < 15; ++kk) {
    const int k = 15*hi + kk;
    w1l[kk] = Wd1[k];
    rkl[kk] = (5.0f/OPN) * (float)(k + 1);
  }

  const float4* xp = reinterpret_cast<const float4*>(x);
  float4 xv = xp[b_r*SEQ + 0];   // x0, dB, x2, x3

  float s1[15];
  float ha = 0.0f;
  #pragma unroll
  for (int kk = 0; kk < 15; ++kk) {
    float sv = xv.y + s0[b_r*OPN + 15*hi + kk];
    sv = fminf(fmaxf(sv, -rkl[kk]), rkl[kk]);
    s1[kk] = sv;
    ha += sv * w1l[kk];
  }
  ha += __shfl_xor(ha, 16, 32);
  float Hh = ha + vdot1 + bd1_0;     // H_hyst0

  // lane pairs hold identical values -> duplicate same-value stores are safe
  hh[r] = Hh; xs0s[r] = xv.x; xs2s[r] = xv.z; xs3s[r] = xv.w;
  varS[r][0]=va0; varS[r][1]=va1; varS[r][2]=va2; varS[r][3]=va3;
  __syncthreads();

  // ---- constant B fragments (held in VGPRs for whole loop) ----
  // b0/b1: columns j0/j1 of Wh2h^T, rows K=30,31 carry Wx2h[:,1]/Wx2h[:,2]
  // b2: column-uniform W_dnn2 (zeros at K>=30) -> D = H_hyst + h2.Wd2
  v16h b0, b1, b2;
  #pragma unroll
  for (int e = 0; e < 16; ++e) {
    const int k = (e < 8) ? (hi*8 + e) : (16 + hi*8 + (e - 8));
    float v0, v1, v2;
    if (k < HID) {
      v0 = Wh2h[j0*HID + k];
      v1 = ok1 ? Wh2h[j1*HID + k] : 0.0f;
      v2 = Wd2[k];
    } else {
      const int c = (k == HID) ? 1 : 2;
      v0 = Wx2h[j0*7 + c];
      v1 = ok1 ? Wx2h[j1*7 + c] : 0.0f;
      v2 = 0.0f;
    }
    b0[e] = (_Float16)v0; b1[e] = (_Float16)v1; b2[e] = (_Float16)v2;
  }

  const float w0a = Wx2h[j0*7 + 0];
  const float w0b = ok1 ? Wx2h[j1*7 + 0] : 0.0f;

  // ---- C var-part fragments (constant over t) ----
  v8f cv0, cv1;
  #pragma unroll
  for (int v = 0; v < 8; ++v) {
    const int m = v + 8*hi;
    float a0c = 0.0f, a1c = 0.0f;
    #pragma unroll
    for (int i = 0; i < NVAR; ++i) {
      const float vv = varS[m][i];
      a0c += vv * Wx2h[j0*7 + 3 + i];
      a1c += vv * (ok1 ? Wx2h[j1*7 + 3 + i] : 0.0f);
    }
    cv0[v] = a0c; cv1[v] = a1c;
  }

  // ---- t = 0 init h2 (D-layout, VALU) ----
  float sumW2 = 0.0f, rsa = 0.0f, rsb = 0.0f;
  #pragma unroll
  for (int k = 0; k < HID; ++k) {
    sumW2 += Wd2[k];
    rsa   += Wh2h[j0*HID + k];
    rsb   += ok1 ? Wh2h[j1*HID + k] : 0.0f;
  }
  const float w1a = Wx2h[j0*7+1], w2a = Wx2h[j0*7+2];
  const float w1b = ok1 ? Wx2h[j1*7+1] : 0.0f;
  const float w2b = ok1 ? Wx2h[j1*7+2] : 0.0f;
  const float inv_sumW2 = __builtin_amdgcn_rcpf(sumW2);

  #pragma unroll
  for (int v = 0; v < 8; ++v) {
    const int m = v + 8*hi;
    const float x0m = xs0s[m], x2m = xs2s[m], x3m = xs3s[m];
    const float kb = (x0m - hh[m]) * inv_sumW2;   // H_eddy_init / sum(W_dnn2)
    const float h0 = sigf(cv0[v] + x0m*w0a + x2m*w1a + x3m*w2a + kb*rsa);
    const float h1 = sigf(cv1[v] + x0m*w0b + x2m*w1b + x3m*w2b + kb*rsb);
    abase[m*32 + j0]       = (_Float16)h0;
    abase[m*t1str + t1off] = (_Float16)h1;   // redirected to dump row if !ok1
  }

  // ---- main recurrence ----
  for (int t = 1; t < SEQ; ++t) {
    xv = xp[b_r*SEQ + t];
    __builtin_prefetch(&xp[b_r*SEQ + t + 1], 0, 3);

    // out_{t-1} C operand: previous-step H_hyst (read hh before overwrite)
    v8f c2;
    #pragma unroll
    for (int v = 0; v < 8; ++v) c2[v] = hh[v + 8*hi] + bd2_0;

    // hysteresis update + dot (split over lane pair)
    float hacc = 0.0f;
    #pragma unroll
    for (int kk = 0; kk < 15; ++kk) {
      float sv = fminf(fmaxf(xv.y + s1[kk], -rkl[kk]), rkl[kk]);
      s1[kk] = sv;
      hacc += sv * w1l[kk];
    }
    hacc += __shfl_xor(hacc, 16, 32);
    Hh = hacc + vdot1 + bd1_0;

    __syncthreads();
    // duplicate same-value stores from lane pairs (no exec-mask churn)
    hh[r]   = Hh;
    xs0s[r] = xv.x;
    Abuf[r][30] = (_Float16)xv.z;  // K=30 slot: x2_t
    Abuf[r][31] = (_Float16)xv.w;  // K=31 slot: x3_t
    __syncthreads();

    // A fragment (16x32 f16): lane row r, K-halves per hi
    const v8h alo = *(const v8h*)(&Abuf[r][8*hi]);
    const v8h ahi = *(const v8h*)(&Abuf[r][16 + 8*hi]);
    v16h a;
    #pragma unroll
    for (int i = 0; i < 8; ++i) { a[i] = alo[i]; a[8+i] = ahi[i]; }

    // C operands: var part + x0 injection
    v8f c0, c1;
    #pragma unroll
    for (int v = 0; v < 8; ++v) {
      const float x0m = xs0s[v + 8*hi];
      c0[v] = cv0[v] + x0m * w0a;
      c1[v] = cv1[v] + x0m * w0b;
    }

    v8f d0 = __builtin_amdgcn_wmma_f32_16x16x32_f16(false, a, false, b0, (short)0, c0, false, false);
    v8f d1 = __builtin_amdgcn_wmma_f32_16x16x32_f16(false, a, false, b1, (short)0, c1, false, false);
    v8f d2 = __builtin_amdgcn_wmma_f32_16x16x32_f16(false, a, false, b2, (short)0, c2, false, false);

    // out_{t-1} = H_hyst_{t-1} + h2_{t-1}.Wd2 + b  (columns uniform; lanes 0/16 store)
    if (((t - 1) >= NINIT) && r == 0) {
      #pragma unroll
      for (int v = 0; v < 8; ++v)
        Hout[(size_t)(base + v + 8*hi) * SOUT + (t - 1 - NINIT)] = d2[v];
    }

    // h2_t = sigmoid(d) -> stage into A buffer for next step (branch-free)
    #pragma unroll
    for (int v = 0; v < 8; ++v) {
      const int m = v + 8*hi;
      abase[m*32 + j0]       = (_Float16)sigf(d0[v]);
      abase[m*t1str + t1off] = (_Float16)sigf(d1[v]);
    }
  }

  // ---- epilogue: out at t = SEQ-1 uses h2_{S-1} ----
  __syncthreads();
  v8f c2;
  #pragma unroll
  for (int v = 0; v < 8; ++v) c2[v] = hh[v + 8*hi] + bd2_0;
  const v8h alo = *(const v8h*)(&Abuf[r][8*hi]);
  const v8h ahi = *(const v8h*)(&Abuf[r][16 + 8*hi]);
  v16h a;
  #pragma unroll
  for (int i = 0; i < 8; ++i) { a[i] = alo[i]; a[8+i] = ahi[i]; }
  v8f d2 = __builtin_amdgcn_wmma_f32_16x16x32_f16(false, a, false, b2, (short)0, c2, false, false);
  if (r == 0) {
    #pragma unroll
    for (int v = 0; v < 8; ++v)
      Hout[(size_t)(base + v + 8*hi) * SOUT + (SOUT - 1)] = d2[v];
  }
}

// ---- tail: trapezoid power integral + tiny MLP; one warp per batch row ----
__global__ __launch_bounds__(256)
void mmi_tail(const float* __restrict__ x,   const float* __restrict__ var,
              const float* __restrict__ amps,const float* __restrict__ Wm1,
              const float* __restrict__ bm1, const float* __restrict__ Wm2,
              const float* __restrict__ bm2, const float* __restrict__ Wm3,
              const float* __restrict__ bm3, const float* __restrict__ Hmat,
              float* __restrict__ out2)
{
  const int warp = threadIdx.x >> 5;
  const int lane = threadIdx.x & 31;
  const int b    = blockIdx.x * 8 + warp;

  const float a0 = amps[b*2+0], a1 = amps[b*2+1];
  const float* Hrow = Hmat + (size_t)b * SOUT;
  const float* xrow = x + ((size_t)b * SEQ + NINIT) * 4;

  float acc = 0.0f;
  for (int i = lane; i < SOUT - 1; i += 32) {
    const float Bm0 = xrow[i*4]       * a0;
    const float Bm1 = xrow[(i+1)*4]   * a0;
    const float Hm0 = Hrow[i]         * a1;
    const float Hm1 = Hrow[i+1]       * a1;
    acc += (Bm1 - Bm0) * (Hm1 + Hm0) * 0.5f;
  }
  #pragma unroll
  for (int off = 16; off > 0; off >>= 1) acc += __shfl_xor(acc, off, 32);

  const float v0 = var[b*4+0], v1 = var[b*4+1], v2 = var[b*4+2], v3 = var[b*4+3];
  float P = acc * __expf((v0*N01 + N00) * 2.302585092994046f); // 10^(...)
  P = fmaxf(P, 1e-12f);
  const float Pcv = (__log10f(P) - N40) * (1.0f / N41);

  const float in5[5] = {v0, v1, v2, v3, Pcv};
  __shared__ float h1s[8][64];
  for (int o = lane; o < 64; o += 32) {
    float s = bm1[o];
    #pragma unroll
    for (int i = 0; i < 5; ++i) s += Wm1[o*5 + i] * in5[i];
    h1s[warp][o] = fmaxf(s, 0.0f);
  }
  __syncthreads();

  float hv = bm2[lane];
  #pragma unroll
  for (int k = 0; k < 64; ++k) hv += Wm2[lane*64 + k] * h1s[warp][k];
  hv = fmaxf(hv, 0.0f);

  float sa = hv * Wm3[lane];
  #pragma unroll
  for (int off = 16; off > 0; off >>= 1) sa += __shfl_xor(sa, off, 32);

  if (lane == 0) out2[b] = Pcv + sa + bm3[0];
}

extern "C" void kernel_launch(void* const* d_in, const int* in_sizes, int n_in,
                              void* d_out, int out_size, void* d_ws, size_t ws_size,
                              hipStream_t stream)
{
  (void)in_sizes; (void)n_in; (void)out_size; (void)d_ws; (void)ws_size;

  const float* x    = (const float*)d_in[0];
  const float* var  = (const float*)d_in[1];
  const float* amps = (const float*)d_in[2];
  const float* s0   = (const float*)d_in[3];
  const float* Wd1  = (const float*)d_in[4];
  const float* bd1  = (const float*)d_in[5];
  const float* Wx2h = (const float*)d_in[6];
  const float* Wh2h = (const float*)d_in[7];
  const float* Wd2  = (const float*)d_in[8];
  const float* bd2  = (const float*)d_in[9];
  const float* Wm1  = (const float*)d_in[10];
  const float* bm1  = (const float*)d_in[11];
  const float* Wm2  = (const float*)d_in[12];
  const float* bm2  = (const float*)d_in[13];
  const float* Wm3  = (const float*)d_in[14];
  const float* bm3  = (const float*)d_in[15];
  // d_in[16] = n_init (16), fixed at compile time

  float* Hout = (float*)d_out;
  float* out2 = Hout + (size_t)BSZ * SOUT;

  mmi_recurrent<<<BSZ/16, 32, 0, stream>>>(x, var, s0, Wd1, bd1, Wx2h, Wh2h, Wd2, bd2, Hout);
  mmi_tail<<<BSZ/8, 256, 0, stream>>>(x, var, amps, Wm1, bm1, Wm2, bm2, Wm3, bm3, Hout, out2);
}
